// HierarchicalTransformerBlock_39857296506975
// MI455X (gfx1250) — matrix operand order repaired and checked by reading, not compiled, not run on hardware
//
#include <hip/hip_runtime.h>
#include <hip/hip_bf16.h>

typedef __attribute__((ext_vector_type(16))) _Float16 v16h;
typedef __attribute__((ext_vector_type(8)))  float    v8f;
typedef _Float16 f16;

#define DEVFN static __device__ __forceinline__

// ---------------- constants ----------------
#define NB    2
#define SS    1024
#define NTOK  2048          // B*S
#define DD    768
#define NHH   12
#define DHH   64
#define HID   1536
#define GG    4
#define EPG   2
#define EE    8

// D = A*B + C, 16x16 tile, K=32, f16 in / f32 acc
DEVFN v8f wmma32(v16h a, v16h b, v8f c) {
  return __builtin_amdgcn_wmma_f32_16x16x32_f16(false, a, false, b, (short)0, c, false, false);
}

// CDNA5 async copy global->LDS (16 bytes per lane), tracked by ASYNCcnt.
DEVFN void async_ld128(unsigned lds_addr, const void* gaddr) {
  asm volatile("global_load_async_to_lds_b128 %0, %1, off"
               :: "v"(lds_addr), "v"(gaddr) : "memory");
}
DEVFN void wait_async0() { asm volatile("s_wait_asynccnt 0x0" ::: "memory"); }
DEVFN void wait_async8() { asm volatile("s_wait_asynccnt 0x8" ::: "memory"); }
DEVFN unsigned lds_off(const void* p) { return (unsigned)(uintptr_t)p; }

// A-matrix fragment (16x32 f16) from row-major tile (ISA 7.12.2).
DEVFN v16h a_frag(const f16* base, int pitch) {
  int lane = threadIdx.x & 31;
  int m = lane & 15, h = lane >> 4;
  const f16* p = base + m * pitch + h * 16;
  v16h a;
  ((float4*)&a)[0] = *(const float4*)(p);
  ((float4*)&a)[1] = *(const float4*)(p + 8);
  return a;
}

// B-matrix fragment (32x16 f16) from a B^T tile (bt[n][k], row-major).
DEVFN v16h bt_frag(const f16* bt, int pitch) {
  int lane = threadIdx.x & 31;
  int n = lane & 15, koff = (lane >> 4) << 4;
  const f16* p = bt + n * pitch + koff;
  v16h b;
  ((float4*)&b)[0] = *(const float4*)(p);
  ((float4*)&b)[1] = *(const float4*)(p + 8);
  return b;
}
// C/D f32 element (m,n): n = lane%16, m = v + 8*(lane/16).

// ---------------- LayerNorm -> f16 ----------------
__global__ void __launch_bounds__(256) k_ln(const float* __restrict__ X,
                                            const float* __restrict__ w,
                                            const float* __restrict__ b,
                                            f16* __restrict__ out) {
  int t = blockIdx.x;
  const float* x = X + (size_t)t * DD;
  __shared__ float red[256];
  float s = 0.f;
  for (int i = threadIdx.x; i < DD; i += 256) s += x[i];
  red[threadIdx.x] = s; __syncthreads();
  for (int o = 128; o > 0; o >>= 1) { if (threadIdx.x < o) red[threadIdx.x] += red[threadIdx.x + o]; __syncthreads(); }
  float mean = red[0] / (float)DD; __syncthreads();
  float vs = 0.f;
  for (int i = threadIdx.x; i < DD; i += 256) { float d = x[i] - mean; vs += d * d; }
  red[threadIdx.x] = vs; __syncthreads();
  for (int o = 128; o > 0; o >>= 1) { if (threadIdx.x < o) red[threadIdx.x] += red[threadIdx.x + o]; __syncthreads(); }
  float rstd = rsqrtf(red[0] / (float)DD + 1e-5f);
  for (int i = threadIdx.x; i < DD; i += 256)
    out[(size_t)t * DD + i] = (f16)((x[i] - mean) * rstd * w[i] + b[i]);
}

// ---------------- QKV projection: [N,768] x [2304,768]^T, 128x64 tile, double-buffered ----------------
__global__ void __launch_bounds__(128) k_gemm_qkv(const f16* __restrict__ A,
                                                  const float* __restrict__ W,
                                                  const float* __restrict__ bias,
                                                  f16* __restrict__ qb, f16* __restrict__ kb,
                                                  f16* __restrict__ vb) {
  __shared__ __align__(16) f16 sA[2][128 * 32];
  __shared__ __align__(16) f16 sB[2][64 * 32];
  int m0 = blockIdx.x * 128, j0 = blockIdx.y * 64;
  int tid = threadIdx.x, w = tid >> 5;
  v8f acc[2][4] = {};
  const char* arow = (const char*)(A + (size_t)(m0 + tid) * DD);
  unsigned ldsA[2] = { lds_off(&sA[0][tid * 32]), lds_off(&sA[1][tid * 32]) };

  auto stageA = [&](int kk, int buf) {
    const char* g = arow + kk * 2;
    #pragma unroll
    for (int i = 0; i < 4; i++) async_ld128(ldsA[buf] + i * 16, g + i * 16);
  };
  auto stageB = [&](int kk, int buf) {                      // sB[j][k] = (f16)W[j0+j][kk+k]
    int r = tid >> 1, part = (tid & 1) * 16;
    const float4* s4 = (const float4*)(W + (size_t)(j0 + r) * DD + kk + part);
    f16* d = &sB[buf][r * 32 + part];
    #pragma unroll
    for (int i = 0; i < 4; i++) {
      float4 f = s4[i];
      d[4*i+0] = (f16)f.x; d[4*i+1] = (f16)f.y; d[4*i+2] = (f16)f.z; d[4*i+3] = (f16)f.w;
    }
  };

  stageA(0, 0); stageB(0, 0);
  wait_async0(); __syncthreads();
  int buf = 0;
  for (int kk = 0; kk < DD; kk += 32, buf ^= 1) {
    int nk = kk + 32;
    if (nk < DD) { stageA(nk, buf ^ 1); stageB(nk, buf ^ 1); }   // prefetch next tile
    v16h bf[4];
    #pragma unroll
    for (int nt = 0; nt < 4; nt++) bf[nt] = bt_frag(&sB[buf][nt * 16 * 32], 32);
    v16h a0 = a_frag(&sA[buf][(w * 32) * 32], 32);
    v16h a1 = a_frag(&sA[buf][(w * 32 + 16) * 32], 32);
    #pragma unroll
    for (int nt = 0; nt < 4; nt++) {
      acc[0][nt] = wmma32(a0, bf[nt], acc[0][nt]);
      acc[1][nt] = wmma32(a1, bf[nt], acc[1][nt]);
    }
    wait_async0();
    __syncthreads();
  }
  int lane = tid & 31, n = lane & 15, mb = (lane >> 4) * 8;
  int which = j0 / DD;                 // uniform: tile lies entirely in q, k or v
  int rb = j0 - which * DD;
  #pragma unroll
  for (int mi = 0; mi < 2; mi++)
    #pragma unroll
    for (int nt = 0; nt < 4; nt++)
      #pragma unroll
      for (int v = 0; v < 8; v++) {
        int m = m0 + w * 32 + mi * 16 + mb + v;
        int r = rb + nt * 16 + n;
        float val = acc[mi][nt][v] + bias[j0 + nt * 16 + n];
        int hh = r >> 6, dh = r & 63;
        int btk = m >> 10, s = m & 1023;
        size_t idx = (((size_t)(btk * NHH + hh)) * SS + s) * DHH + dh;
        if (which == 0)      qb[idx] = (f16)(val * 0.125f);  // fold 1/sqrt(DH)
        else if (which == 1) kb[idx] = (f16)val;
        else                 vb[idx] = (f16)val;
      }
}

// ---------------- Flash attention: 1 wave per (bh, 16 q-rows), K double-buffered ----------------
__global__ void __launch_bounds__(32) k_attn(const f16* __restrict__ Q, const f16* __restrict__ K,
                                             const f16* __restrict__ V, f16* __restrict__ O) {
  __shared__ __align__(16) f16 sK[2][32 * 64];
  __shared__ __align__(16) f16 sV[32 * 64];
  __shared__ __align__(16) f16 sP[16 * 32];
  int q0 = blockIdx.x * 16, bh = blockIdx.y;
  const f16* qp = Q + ((size_t)bh * SS + q0) * DHH;
  const f16* kp = K + (size_t)bh * SS * DHH;
  const f16* vp = V + (size_t)bh * SS * DHH;
  int lane = threadIdx.x;
  unsigned ldsK[2] = { lds_off(&sK[0][lane * 64]), lds_off(&sK[1][lane * 64]) };
  unsigned ldsV = lds_off(&sV[lane * 64]);

  auto stage = [&](const f16* base, int c, unsigned lds) {  // copy rows c..c+31 (128B/lane)
    const char* g = (const char*)(base + (size_t)(c + lane) * DHH);
    #pragma unroll
    for (int i = 0; i < 8; i++) async_ld128(lds + i * 16, g + i * 16);
  };

  v16h aq0 = a_frag(qp, DHH);        // DH 0..31
  v16h aq1 = a_frag(qp + 32, DHH);   // DH 32..63
  v8f o[4] = {};
  float run_m[8], run_l[8];
  #pragma unroll
  for (int v = 0; v < 8; v++) { run_m[v] = -1e30f; run_l[v] = 0.f; }

  stage(kp, 0, ldsK[0]);
  int buf = 0;
  for (int c = 0; c < SS; c += 32, buf ^= 1) {
    wait_async0();                                          // K_c (and any stragglers) resident
    __syncthreads();
    v8f sc[2] = {};
    #pragma unroll
    for (int half = 0; half < 2; half++) {                 // keys c+half*16 .. +15
      const f16* bt = &sK[buf][half * 16 * 64];            // B^T[n][k] = K[key n][k]
      sc[half] = wmma32(aq0, bt_frag(bt, 64), sc[half]);
      sc[half] = wmma32(aq1, bt_frag(bt + 32, 64), sc[half]);
    }
    // prefetch V_c now, K_{c+32} behind it (async loads complete in order)
    stage(vp, c, ldsV);
    bool moreK = (c + 32 < SS);
    if (moreK) stage(kp, c + 32, ldsK[buf ^ 1]);
    float alpha[8];
    #pragma unroll
    for (int v = 0; v < 8; v++) {                          // online softmax, rows in 16-lane halves
      float mx = fmaxf(sc[0][v], sc[1][v]);
      #pragma unroll
      for (int t = 8; t; t >>= 1) mx = fmaxf(mx, __shfl_xor(mx, t, 32));
      float mnew = fmaxf(run_m[v], mx);
      alpha[v] = __expf(run_m[v] - mnew);
      float p0 = __expf(sc[0][v] - mnew), p1 = __expf(sc[1][v] - mnew);
      sc[0][v] = p0; sc[1][v] = p1;
      float rs = p0 + p1;
      #pragma unroll
      for (int t = 8; t; t >>= 1) rs += __shfl_xor(rs, t, 32);
      run_l[v] = run_l[v] * alpha[v] + rs;
      run_m[v] = mnew;
    }
    { int n = lane & 15, hi = lane >> 4;                   // C-layout -> A-layout via LDS
      #pragma unroll
      for (int v = 0; v < 8; v++) {
        sP[(v + 8 * hi) * 32 + n]      = (f16)sc[0][v];
        sP[(v + 8 * hi) * 32 + 16 + n] = (f16)sc[1][v];
      } }
    if (moreK) wait_async8(); else wait_async0();          // V done; K' may stay in flight
    __syncthreads();
    v16h pf = a_frag(sP, 32);
    #pragma unroll
    for (int nt = 0; nt < 4; nt++) {
      #pragma unroll
      for (int v = 0; v < 8; v++) o[nt][v] *= alpha[v];
      int n = lane & 15, koff = (lane >> 4) * 16;          // B frag from V rows [32][64]
      v16h bv;
      #pragma unroll
      for (int i = 0; i < 16; i++) bv[i] = sV[(koff + i) * 64 + nt * 16 + n];
      o[nt] = wmma32(pf, bv, o[nt]);
    }
    __syncthreads();
  }
  int bt = bh / NHH, hh = bh % NHH;
  int n = lane & 15, mb = (lane >> 4) * 8;
  #pragma unroll
  for (int nt = 0; nt < 4; nt++)
    #pragma unroll
    for (int v = 0; v < 8; v++) {
      int m = q0 + mb + v;
      size_t idx = ((size_t)(bt * SS + m)) * DD + hh * DHH + nt * 16 + n;
      O[idx] = (f16)(o[nt][v] / run_l[v]);
    }
}

// ---------------- out-proj + residual -> x2 (f32), 128x64 tile, double-buffered ----------------
__global__ void __launch_bounds__(128) k_gemm_out(const f16* __restrict__ A,
                                                  const float* __restrict__ W,
                                                  const float* __restrict__ bias,
                                                  const float* __restrict__ Xin,
                                                  float* __restrict__ X2) {
  __shared__ __align__(16) f16 sA[2][128 * 32];
  __shared__ __align__(16) f16 sB[2][64 * 32];
  int m0 = blockIdx.x * 128, j0 = blockIdx.y * 64;
  int tid = threadIdx.x, w = tid >> 5;
  v8f acc[2][4] = {};
  const char* arow = (const char*)(A + (size_t)(m0 + tid) * DD);
  unsigned ldsA[2] = { lds_off(&sA[0][tid * 32]), lds_off(&sA[1][tid * 32]) };

  auto stageA = [&](int kk, int buf) {
    const char* g = arow + kk * 2;
    #pragma unroll
    for (int i = 0; i < 4; i++) async_ld128(ldsA[buf] + i * 16, g + i * 16);
  };
  auto stageB = [&](int kk, int buf) {
    int r = tid >> 1, part = (tid & 1) * 16;
    const float4* s4 = (const float4*)(W + (size_t)(j0 + r) * DD + kk + part);
    f16* d = &sB[buf][r * 32 + part];
    #pragma unroll
    for (int i = 0; i < 4; i++) {
      float4 f = s4[i];
      d[4*i+0] = (f16)f.x; d[4*i+1] = (f16)f.y; d[4*i+2] = (f16)f.z; d[4*i+3] = (f16)f.w;
    }
  };

  stageA(0, 0); stageB(0, 0);
  wait_async0(); __syncthreads();
  int buf = 0;
  for (int kk = 0; kk < DD; kk += 32, buf ^= 1) {
    int nk = kk + 32;
    if (nk < DD) { stageA(nk, buf ^ 1); stageB(nk, buf ^ 1); }
    v16h bf[4];
    #pragma unroll
    for (int nt = 0; nt < 4; nt++) bf[nt] = bt_frag(&sB[buf][nt * 16 * 32], 32);
    v16h a0 = a_frag(&sA[buf][(w * 32) * 32], 32);
    v16h a1 = a_frag(&sA[buf][(w * 32 + 16) * 32], 32);
    #pragma unroll
    for (int nt = 0; nt < 4; nt++) {
      acc[0][nt] = wmma32(a0, bf[nt], acc[0][nt]);
      acc[1][nt] = wmma32(a1, bf[nt], acc[1][nt]);
    }
    wait_async0();
    __syncthreads();
  }
  int lane = tid & 31, n = lane & 15, mb = (lane >> 4) * 8;
  #pragma unroll
  for (int mi = 0; mi < 2; mi++)
    #pragma unroll
    for (int nt = 0; nt < 4; nt++)
      #pragma unroll
      for (int v = 0; v < 8; v++) {
        int m = m0 + w * 32 + mi * 16 + mb + v;
        int j = j0 + nt * 16 + n;
        X2[(size_t)m * DD + j] = Xin[(size_t)m * DD + j] + acc[mi][nt][v] + bias[j];
      }
}

// ---------------- hierarchical router: 1 wave per token ----------------
__global__ void __launch_bounds__(32) k_router(const f16* __restrict__ H,
                                               const float* __restrict__ Wg, const float* __restrict__ bg,
                                               const float* __restrict__ We, const float* __restrict__ be,
                                               int* __restrict__ flat, float* __restrict__ prob) {
  int t = blockIdx.x, lane = threadIdx.x;
  const f16* h = H + (size_t)t * DD;
  float gl[GG], el[EE];
  #pragma unroll
  for (int g = 0; g < GG; g++) {
    float s = 0.f;
    for (int d = lane; d < DD; d += 32) s += (float)h[d] * Wg[d * GG + g];
    #pragma unroll
    for (int o = 16; o; o >>= 1) s += __shfl_xor(s, o, 32);
    gl[g] = s + bg[g];
  }
  #pragma unroll
  for (int g = 0; g < GG; g++)
    #pragma unroll
    for (int e = 0; e < EPG; e++) {
      float s = 0.f;
      const float* wv = We + ((size_t)g * DD) * EPG + e;
      for (int d = lane; d < DD; d += 32) s += (float)h[d] * wv[d * EPG];
      #pragma unroll
      for (int o = 16; o; o >>= 1) s += __shfl_xor(s, o, 32);
      el[g * EPG + e] = s + be[g * EPG + e];
    }
  float mx = gl[0]; int gi = 0;
  #pragma unroll
  for (int g = 1; g < GG; g++) if (gl[g] > mx) { mx = gl[g]; gi = g; }
  float den = 0.f;
  #pragma unroll
  for (int g = 0; g < GG; g++) den += __expf(gl[g] - mx);
  float pg = 1.0f / den;                                    // exp(0)/den at argmax
  float e0 = el[gi * EPG], e1 = el[gi * EPG + 1];
  float em = fmaxf(e0, e1);
  float d0 = __expf(e0 - em), d1 = __expf(e1 - em);
  int ei = (e1 > e0) ? 1 : 0;
  float pe = (ei ? d1 : d0) / (d0 + d1);
  if (lane == 0) { flat[t] = gi * EPG + ei; prob[t] = pg * pe; }
}

__global__ void k_zero(int* counts) { if (threadIdx.x < EE) counts[threadIdx.x] = 0; }

__global__ void __launch_bounds__(256) k_dispatch(const int* __restrict__ flat,
                                                  int* __restrict__ counts,
                                                  int* __restrict__ tlist) {
  int t = blockIdx.x * 256 + threadIdx.x;
  if (t >= NTOK) return;
  int e = flat[t];
  int slot = atomicAdd(&counts[e], 1);
  tlist[e * NTOK + slot] = t;
}

// ---------------- MoE GEMM1: gathered rows x W1[e] -> gelu -> hid f16 ----------------
__global__ void __launch_bounds__(128) k_moe1(const f16* __restrict__ H, const float* __restrict__ W1,
                                              const float* __restrict__ b1,
                                              const int* __restrict__ counts, const int* __restrict__ tlist,
                                              f16* __restrict__ hid) {
  int e = blockIdx.z;
  int cnt = counts[e];
  int m0 = blockIdx.x * 128;
  if (m0 >= cnt) return;                     // uniform across block: safe
  int h0 = blockIdx.y * 64;
  __shared__ __align__(16) f16 sA[2][128 * 32];
  __shared__ __align__(16) f16 sB[2][64 * 32];
  __shared__ int rowTok[128];
  int tid = threadIdx.x, w = tid >> 5;
  { int m = m0 + tid;
    rowTok[tid] = (m < cnt) ? tlist[e * NTOK + m] : tlist[e * NTOK]; }   // clamp to valid row
  __syncthreads();
  v8f acc[2][4] = {};
  const float* Wb = W1 + (size_t)e * DD * HID;
  const char* arow = (const char*)(H + (size_t)rowTok[tid] * DD);
  unsigned ldsA[2] = { lds_off(&sA[0][tid * 32]), lds_off(&sA[1][tid * 32]) };

  auto stageA = [&](int kk, int buf) {
    const char* g = arow + kk * 2;
    #pragma unroll
    for (int i = 0; i < 4; i++) async_ld128(ldsA[buf] + i * 16, g + i * 16);
  };
  auto stageB = [&](int kk, int buf) {                      // sB[h][d] = W1[e][kk+d][h0+h]
    int d_ = tid >> 2, sub = (tid & 3) * 16;
    const float4* s4 = (const float4*)(Wb + (size_t)(kk + d_) * HID + h0 + sub);
    #pragma unroll
    for (int i = 0; i < 4; i++) {
      float4 f = s4[i];
      f16* dst = &sB[buf][(sub + 4 * i) * 32 + d_];
      dst[0]  = (f16)f.x; dst[32] = (f16)f.y; dst[64] = (f16)f.z; dst[96] = (f16)f.w;
    }
  };

  stageA(0, 0); stageB(0, 0);
  wait_async0(); __syncthreads();
  int buf = 0;
  for (int kk = 0; kk < DD; kk += 32, buf ^= 1) {
    int nk = kk + 32;
    if (nk < DD) { stageA(nk, buf ^ 1); stageB(nk, buf ^ 1); }
    v16h bf[4];
    #pragma unroll
    for (int nt = 0; nt < 4; nt++) bf[nt] = bt_frag(&sB[buf][nt * 16 * 32], 32);
    v16h a0 = a_frag(&sA[buf][(w * 32) * 32], 32);
    v16h a1 = a_frag(&sA[buf][(w * 32 + 16) * 32], 32);
    #pragma unroll
    for (int nt = 0; nt < 4; nt++) {
      acc[0][nt] = wmma32(a0, bf[nt], acc[0][nt]);
      acc[1][nt] = wmma32(a1, bf[nt], acc[1][nt]);
    }
    wait_async0();
    __syncthreads();
  }
  int lane = tid & 31, n = lane & 15, mb = (lane >> 4) * 8;
  #pragma unroll
  for (int mi = 0; mi < 2; mi++)
    #pragma unroll
    for (int nt = 0; nt < 4; nt++)
      #pragma unroll
      for (int v = 0; v < 8; v++) {
        int m = m0 + w * 32 + mi * 16 + mb + v;
        if (m >= cnt) continue;
        int t = rowTok[w * 32 + mi * 16 + mb + v];
        int hcol = h0 + nt * 16 + n;
        float xv = acc[mi][nt][v] + b1[e * HID + hcol];
        float gv = 0.5f * xv * (1.0f + tanhf(0.7978845608028654f * (xv + 0.044715f * xv * xv * xv)));
        hid[(size_t)t * HID + hcol] = (f16)gv;
      }
}

// ---------------- MoE GEMM2: hid x W2[e] -> prob*y + x2 -> out ----------------
__global__ void __launch_bounds__(128) k_moe2(const f16* __restrict__ Hid, const float* __restrict__ W2,
                                              const float* __restrict__ b2,
                                              const int* __restrict__ counts, const int* __restrict__ tlist,
                                              const float* __restrict__ prob,
                                              const float* __restrict__ X2, float* __restrict__ Out) {
  int e = blockIdx.z;
  int cnt = counts[e];
  int m0 = blockIdx.x * 128;
  if (m0 >= cnt) return;
  int d0 = blockIdx.y * 64;
  __shared__ __align__(16) f16 sA[2][128 * 32];
  __shared__ __align__(16) f16 sB[2][64 * 32];
  __shared__ int rowTok[128];
  int tid = threadIdx.x, w = tid >> 5;
  { int m = m0 + tid;
    rowTok[tid] = (m < cnt) ? tlist[e * NTOK + m] : tlist[e * NTOK]; }
  __syncthreads();
  v8f acc[2][4] = {};
  const float* Wb = W2 + (size_t)e * HID * DD;
  const char* arow = (const char*)(Hid + (size_t)rowTok[tid] * HID);
  unsigned ldsA[2] = { lds_off(&sA[0][tid * 32]), lds_off(&sA[1][tid * 32]) };

  auto stageA = [&](int kk, int buf) {
    const char* g = arow + kk * 2;
    #pragma unroll
    for (int i = 0; i < 4; i++) async_ld128(ldsA[buf] + i * 16, g + i * 16);
  };
  auto stageB = [&](int kk, int buf) {                      // sB[d][h] = W2[e][kk+h][d0+d]
    int h_ = tid >> 2, sub = (tid & 3) * 16;
    const float4* s4 = (const float4*)(Wb + (size_t)(kk + h_) * DD + d0 + sub);
    #pragma unroll
    for (int i = 0; i < 4; i++) {
      float4 f = s4[i];
      f16* dst = &sB[buf][(sub + 4 * i) * 32 + h_];
      dst[0]  = (f16)f.x; dst[32] = (f16)f.y; dst[64] = (f16)f.z; dst[96] = (f16)f.w;
    }
  };

  stageA(0, 0); stageB(0, 0);
  wait_async0(); __syncthreads();
  int buf = 0;
  for (int kk = 0; kk < HID; kk += 32, buf ^= 1) {
    int nk = kk + 32;
    if (nk < HID) { stageA(nk, buf ^ 1); stageB(nk, buf ^ 1); }
    v16h bf[4];
    #pragma unroll
    for (int nt = 0; nt < 4; nt++) bf[nt] = bt_frag(&sB[buf][nt * 16 * 32], 32);
    v16h a0 = a_frag(&sA[buf][(w * 32) * 32], 32);
    v16h a1 = a_frag(&sA[buf][(w * 32 + 16) * 32], 32);
    #pragma unroll
    for (int nt = 0; nt < 4; nt++) {
      acc[0][nt] = wmma32(a0, bf[nt], acc[0][nt]);
      acc[1][nt] = wmma32(a1, bf[nt], acc[1][nt]);
    }
    wait_async0();
    __syncthreads();
  }
  int lane = tid & 31, n = lane & 15, mb = (lane >> 4) * 8;
  #pragma unroll
  for (int mi = 0; mi < 2; mi++)
    #pragma unroll
    for (int nt = 0; nt < 4; nt++)
      #pragma unroll
      for (int v = 0; v < 8; v++) {
        int m = m0 + w * 32 + mi * 16 + mb + v;
        if (m >= cnt) continue;
        int t = rowTok[w * 32 + mi * 16 + mb + v];
        int dcol = d0 + nt * 16 + n;
        float y = acc[mi][nt][v] + b2[e * DD + dcol];
        Out[(size_t)t * DD + dcol] = X2[(size_t)t * DD + dcol] + prob[t] * y;
      }
}

// ---------------- launch ----------------
extern "C" void kernel_launch(void* const* d_in, const int* in_sizes, int n_in,
                              void* d_out, int out_size, void* d_ws, size_t ws_size,
                              hipStream_t stream) {
  const float* x         = (const float*)d_in[0];
  const float* ln1_w     = (const float*)d_in[1];
  const float* ln1_b     = (const float*)d_in[2];
  const float* in_proj_w = (const float*)d_in[3];
  const float* in_proj_b = (const float*)d_in[4];
  const float* out_proj_w= (const float*)d_in[5];
  const float* out_proj_b= (const float*)d_in[6];
  const float* ln2_w     = (const float*)d_in[7];
  const float* ln2_b     = (const float*)d_in[8];
  const float* Wg        = (const float*)d_in[9];
  const float* bg        = (const float*)d_in[10];
  const float* We        = (const float*)d_in[11];
  const float* be        = (const float*)d_in[12];
  const float* W1        = (const float*)d_in[13];
  const float* b1        = (const float*)d_in[14];
  const float* W2        = (const float*)d_in[15];
  const float* b2        = (const float*)d_in[16];
  float* out = (float*)d_out;

  char* p = (char*)d_ws;
  auto take = [&](size_t bytes) { char* r = p; p += (bytes + 255) & ~(size_t)255; return r; };
  f16*   h1   = (f16*)  take((size_t)NTOK * DD * 2);
  f16*   qb   = (f16*)  take((size_t)NTOK * DD * 2);
  f16*   kb   = (f16*)  take((size_t)NTOK * DD * 2);
  f16*   vb   = (f16*)  take((size_t)NTOK * DD * 2);
  f16*   attn = (f16*)  take((size_t)NTOK * DD * 2);
  float* x2   = (float*)take((size_t)NTOK * DD * 4);
  f16*   h2   = (f16*)  take((size_t)NTOK * DD * 2);
  f16*   hid  = (f16*)  take((size_t)NTOK * HID * 2);
  int*   flat = (int*)  take((size_t)NTOK * 4);
  float* prob = (float*)take((size_t)NTOK * 4);
  int*   cnts = (int*)  take(64);
  int*   tlist= (int*)  take((size_t)EE * NTOK * 4);

  k_ln      <<<NTOK, 256, 0, stream>>>(x, ln1_w, ln1_b, h1);
  k_gemm_qkv<<<dim3(NTOK / 128, (3 * DD) / 64), 128, 0, stream>>>(h1, in_proj_w, in_proj_b, qb, kb, vb);
  k_attn    <<<dim3(SS / 16, NB * NHH), 32, 0, stream>>>(qb, kb, vb, attn);
  k_gemm_out<<<dim3(NTOK / 128, DD / 64), 128, 0, stream>>>(attn, out_proj_w, out_proj_b, x, x2);
  k_ln      <<<NTOK, 256, 0, stream>>>(x2, ln2_w, ln2_b, h2);
  k_zero    <<<1, 32, 0, stream>>>(cnts);
  k_router  <<<NTOK, 32, 0, stream>>>(h2, Wg, bg, We, be, flat, prob);
  k_dispatch<<<NTOK / 256, 256, 0, stream>>>(flat, cnts, tlist);
  k_moe1    <<<dim3(NTOK / 128, HID / 64, EE), 128, 0, stream>>>(h2, W1, b1, cnts, tlist, hid);
  k_moe2    <<<dim3(NTOK / 128, DD / 64, EE), 128, 0, stream>>>(hid, W2, b2, cnts, tlist, prob, x2, out);
}
